// GCNGraph_10333691314776
// MI455X (gfx1250) — matrix operand order repaired and checked by reading, not compiled, not run on hardware
//
#include <hip/hip_runtime.h>
#include <hip/hip_bf16.h>
#include <math.h>

// GCN: 3x GraphConv(norm='both', edge weights) + mean pool + MLP head.
// Dense projections: v_wmma_f32_16x16x32_f16 (f16 in, f32 accum),
// fragments fed by ds_load_b128 from LDS (W stored transposed).

typedef __attribute__((ext_vector_type(16))) _Float16 v16h;
typedef __attribute__((ext_vector_type(8)))  _Float16 v8h;
typedef __attribute__((ext_vector_type(4)))  _Float16 v4h;
typedef __attribute__((ext_vector_type(8)))  float    v8f;

#define NNODES 100000
#define NEDGES 1600000
#define NGRAPH 256
#define FIN    128
#define FH     64

// ---------------- utility kernels ----------------

__global__ void k_zero(float* __restrict__ p, int n) {
  int i = blockIdx.x * blockDim.x + threadIdx.x;
  if (i < n) p[i] = 0.f;
}

__global__ void k_deg(const int* __restrict__ src, const int* __restrict__ dst,
                      float* __restrict__ deg_out, float* __restrict__ deg_in, int e) {
  int i = blockIdx.x * blockDim.x + threadIdx.x;
  if (i < e) {
    atomicAdd(&deg_out[src[i]], 1.f);
    atomicAdd(&deg_in[dst[i]], 1.f);
  }
}

__global__ void k_rsqrt_clamp(float* __restrict__ d, int n) {
  int i = blockIdx.x * blockDim.x + threadIdx.x;
  if (i < n) d[i] = rsqrtf(fmaxf(d[i], 1.f));
}

// a16[i,k] = (f16)(x[i,k] * ns[i]); processes 4 elems/thread
__global__ void k_scale_f16(const float* __restrict__ x, const float* __restrict__ ns,
                            _Float16* __restrict__ a, int n, int K) {
  int i = blockIdx.x * blockDim.x + threadIdx.x;
  int per4 = K >> 2;
  if (i < n * per4) {
    int r = i / per4;
    int c = (i - r * per4) * 4;
    const float4 v = *(const float4*)&x[(size_t)r * K + c];
    float s = ns[r];
    v4h h = {(_Float16)(v.x * s), (_Float16)(v.y * s),
             (_Float16)(v.z * s), (_Float16)(v.w * s)};
    *(v4h*)&a[(size_t)r * K + c] = h;
  }
}

__global__ void k_cvt_f16(const float* __restrict__ w, _Float16* __restrict__ o, int n) {
  int i = blockIdx.x * blockDim.x + threadIdx.x;
  if (i < n) o[i] = (_Float16)w[i];
}

// ---------------- WMMA GEMM: Y[n x 64] = A[n x K] @ B[K x 64] ----------------
// 128 threads = 4 wave32. Block covers 64 rows (4 row-tiles); wave w owns
// col tile [16w,16w+16) and iterates row tiles with acc[4]. W tile stored
// transposed in LDS so B fragments are contiguous (2x ds_load_b128).

__global__ void k_gemm_wmma(const _Float16* __restrict__ A, const _Float16* __restrict__ B,
                            float* __restrict__ Y, int n, int K) {
  __shared__ _Float16 sA[64 * FIN];    // 16 KB max (row-major [r][k])
  __shared__ _Float16 sBt[FH * FIN];   // 16 KB (transposed [col][k])

  const int tid = threadIdx.x;
  const int rowbase = blockIdx.x * 64;

  for (int i = tid; i < K * FH; i += 128) {       // B is [k][col] row-major
    int k = i >> 6, col = i & 63;
    sBt[col * K + k] = B[i];
  }
  for (int i = tid; i < 64 * K; i += 128) {
    int r = i / K, c = i - r * K;
    int row = rowbase + r;
    sA[i] = (row < n) ? A[(size_t)row * K + c] : (_Float16)0.f;
  }
  __syncthreads();

  const int lane = tid & 31;
  const int hi   = lane >> 4;            // lane half
  const int l15  = lane & 15;
  const int colbase = (tid >> 5) * 16;
  const int aoff0 = hi ? 16 : 0;         // A frag K-chunk 0 (k = 16h + s)
  const int aoff1 = hi ? 24 : 8;         // A frag K-chunk 1
  const int boff  = hi ? 16 : 0;         // B frag K-offset

  v8f acc[4] = {v8f{}, v8f{}, v8f{}, v8f{}};

  for (int kt = 0; kt < K; kt += 32) {
    // B fragment: 16 contiguous halfs (32B) in transposed tile
    const v8h* pb = (const v8h*)&sBt[(colbase + l15) * K + kt + boff];
    v8h b0 = pb[0], b1 = pb[1];
    v16h b = __builtin_shufflevector(b0, b1, 0, 1, 2, 3, 4, 5, 6, 7,
                                     8, 9, 10, 11, 12, 13, 14, 15);
#pragma unroll
    for (int rt = 0; rt < 4; ++rt) {
      const _Float16* ar = &sA[(rt * 16 + l15) * K + kt];
      v8h a0 = *(const v8h*)(ar + aoff0);
      v8h a1 = *(const v8h*)(ar + aoff1);
      v16h a = __builtin_shufflevector(a0, a1, 0, 1, 2, 3, 4, 5, 6, 7,
                                       8, 9, 10, 11, 12, 13, 14, 15);
      acc[rt] = __builtin_amdgcn_wmma_f32_16x16x32_f16(
          /*neg_a=*/false, a, /*neg_b=*/false, b,
          /*c_mod=*/(short)0, acc[rt], /*reuse_a=*/false, /*reuse_b=*/false);
    }
  }

  // C/D layout: VGPR r -> M = r (lanes 0-15) or r+8 (lanes 16-31); N = l15.
#pragma unroll
  for (int rt = 0; rt < 4; ++rt) {
#pragma unroll
    for (int r = 0; r < 8; ++r) {
      int row = rowbase + rt * 16 + (hi ? r + 8 : r);
      if (row < n) Y[(size_t)row * FH + colbase + l15] = acc[rt][r];
    }
  }
}

// ---------------- edge scatter: agg[dst] += Y[src] * ew (4 feats/thread) ----------------

__global__ void k_scatter(const int* __restrict__ src, const int* __restrict__ dst,
                          const float* __restrict__ ew, const float* __restrict__ Y,
                          float* __restrict__ agg, int e) {
  unsigned i = blockIdx.x * blockDim.x + threadIdx.x;
  unsigned ed = i >> 4, q = (i & 15) * 4;
  if (ed < (unsigned)e) {
    int s = src[ed], d = dst[ed];
    float w = ew[ed];
    if (q == 0) __builtin_prefetch(&Y[(size_t)s * FH], 0, 1);  // global_prefetch_b8
    const float4 v = *(const float4*)&Y[(size_t)s * FH + q];
    float* ap = &agg[(size_t)d * FH + q];
    atomicAdd(ap + 0, v.x * w);
    atomicAdd(ap + 1, v.y * w);
    atomicAdd(ap + 2, v.z * w);
    atomicAdd(ap + 3, v.w * w);
  }
}

// ---------------- node update: h = relu(agg*nd + b); optionally stage f16(h*ns) ----------------

__global__ void k_node(const float* __restrict__ agg, const float* __restrict__ nd,
                       const float* __restrict__ b, const float* __restrict__ ns,
                       float* __restrict__ h32, _Float16* __restrict__ a16,
                       int n, int writeF16) {
  int i = blockIdx.x * blockDim.x + threadIdx.x;
  if (i < n * (FH / 4)) {
    int r = i >> 4, q = (i & 15) * 4;
    const float4 a = *(const float4*)&agg[(size_t)r * FH + q];
    const float4 bb = *(const float4*)&b[q];
    float dn = nd[r];
    float4 v;
    v.x = fmaxf(a.x * dn + bb.x, 0.f);
    v.y = fmaxf(a.y * dn + bb.y, 0.f);
    v.z = fmaxf(a.z * dn + bb.z, 0.f);
    v.w = fmaxf(a.w * dn + bb.w, 0.f);
    *(float4*)&h32[(size_t)r * FH + q] = v;
    if (writeF16) {
      float s = ns[r];
      v4h h = {(_Float16)(v.x * s), (_Float16)(v.y * s),
               (_Float16)(v.z * s), (_Float16)(v.w * s)};
      *(v4h*)&a16[(size_t)r * FH + q] = h;
    }
  }
}

// ---------------- per-graph mean pooling (4 feats/thread) ----------------

__global__ void k_pool(const float* __restrict__ h, const int* __restrict__ gid,
                       float* __restrict__ pooled, float* __restrict__ counts, int n) {
  int i = blockIdx.x * blockDim.x + threadIdx.x;
  if (i < n * (FH / 4)) {
    int r = i >> 4, q = (i & 15) * 4;
    int g = gid[r];
    const float4 v = *(const float4*)&h[(size_t)r * FH + q];
    float* pp = &pooled[(size_t)g * FH + q];
    atomicAdd(pp + 0, v.x);
    atomicAdd(pp + 1, v.y);
    atomicAdd(pp + 2, v.z);
    atomicAdd(pp + 3, v.w);
    if (q == 0) atomicAdd(&counts[g], 1.f);
  }
}

// ---------------- MLP head: 64 -> 16 -> 8 -> 1, sigmoid ----------------

__global__ void k_mlp(const float* __restrict__ pooled, const float* __restrict__ counts,
                      const float* __restrict__ Dw1, const float* __restrict__ Db1,
                      const float* __restrict__ Dw2, const float* __restrict__ Db2,
                      const float* __restrict__ Dw3, const float* __restrict__ Db3,
                      float* __restrict__ out, int g) {
  int i = threadIdx.x;
  if (i >= g) return;
  float inv = 1.f / fmaxf(counts[i], 1.f);
  float p[FH];
  for (int f = 0; f < FH; ++f) p[f] = pooled[i * FH + f] * inv;
  float h1[16];
  for (int o = 0; o < 16; ++o) {
    float s = Db1[o];
    for (int f = 0; f < FH; ++f) s += p[f] * Dw1[f * 16 + o];
    h1[o] = fmaxf(s, 0.f);
  }
  float h2[8];
  for (int o = 0; o < 8; ++o) {
    float s = Db2[o];
    for (int f = 0; f < 16; ++f) s += h1[f] * Dw2[f * 8 + o];
    h2[o] = fmaxf(s, 0.f);
  }
  float z = Db3[0];
  for (int f = 0; f < 8; ++f) z += h2[f] * Dw3[f];
  out[i] = 1.f / (1.f + expf(-z));
}

// ---------------- launch ----------------

extern "C" void kernel_launch(void* const* d_in, const int* in_sizes, int n_in,
                              void* d_out, int out_size, void* d_ws, size_t ws_size,
                              hipStream_t stream) {
  const float* x   = (const float*)d_in[0];
  const int*   src = (const int*)d_in[1];
  const int*   dst = (const int*)d_in[2];
  const float* ew  = (const float*)d_in[3];
  const int*   gid = (const int*)d_in[4];
  const float* W1  = (const float*)d_in[5];
  const float* b1  = (const float*)d_in[6];
  const float* W2  = (const float*)d_in[7];
  const float* b2  = (const float*)d_in[8];
  const float* W3  = (const float*)d_in[9];
  const float* b3  = (const float*)d_in[10];
  const float* Dw1 = (const float*)d_in[11];
  const float* Db1 = (const float*)d_in[12];
  const float* Dw2 = (const float*)d_in[13];
  const float* Db2 = (const float*)d_in[14];
  const float* Dw3 = (const float*)d_in[15];
  const float* Db3 = (const float*)d_in[16];
  float* out = (float*)d_out;

  const int N = NNODES, E = NEDGES, G = NGRAPH;

  // workspace carve-up (floats); tail kept 16B-aligned for vector f16 stores
  float* ws     = (float*)d_ws;
  float* ns     = ws;                         // N
  float* nd     = ns + N;                     // N
  float* Y      = nd + N;                     // N*64
  float* AGG    = Y + (size_t)N * FH;         // N*64
  float* pooled = AGG + (size_t)N * FH;       // G*64
  float* counts = pooled + (size_t)G * FH;    // G
  _Float16* A16 = (_Float16*)(counts + G);    // N*128 halfs
  _Float16* W16 = A16 + (size_t)N * FIN;      // 128*64 halfs

  const int TB = 256;

  // degrees -> ns, nd
  k_zero<<<(2 * N + TB - 1) / TB, TB, 0, stream>>>(ns, 2 * N);
  k_deg<<<(E + TB - 1) / TB, TB, 0, stream>>>(src, dst, ns, nd, E);
  k_rsqrt_clamp<<<(N + TB - 1) / TB, TB, 0, stream>>>(ns, N);
  k_rsqrt_clamp<<<(N + TB - 1) / TB, TB, 0, stream>>>(nd, N);

  // stage x * ns as f16
  k_scale_f16<<<(N * (FIN / 4) + TB - 1) / TB, TB, 0, stream>>>(x, ns, A16, N, FIN);

  auto layer = [&](int K, const float* W, const float* b, int writeF16) {
    k_cvt_f16<<<(K * FH + TB - 1) / TB, TB, 0, stream>>>(W, W16, K * FH);
    k_gemm_wmma<<<(N + 63) / 64, 128, 0, stream>>>(A16, W16, Y, N, K);
    k_zero<<<(N * FH + TB - 1) / TB, TB, 0, stream>>>(AGG, N * FH);
    k_scatter<<<(E * (FH / 4) + TB - 1) / TB, TB, 0, stream>>>(src, dst, ew, Y, AGG, E);
    k_node<<<(N * (FH / 4) + TB - 1) / TB, TB, 0, stream>>>(AGG, nd, b, ns, Y, A16, N, writeF16);
  };

  layer(FIN, W1, b1, 1);   // 128 -> 64
  layer(FH,  W2, b2, 1);   // 64  -> 64
  layer(FH,  W3, b3, 0);   // 64  -> 64 (final features stay f32 in Y)

  // mean pooling + MLP head
  k_zero<<<(G * FH + G + TB - 1) / TB, TB, 0, stream>>>(pooled, G * FH + G);
  k_pool<<<(N * (FH / 4) + TB - 1) / TB, TB, 0, stream>>>(Y, gid, pooled, counts, N);
  k_mlp<<<1, G, 0, stream>>>(pooled, counts, Dw1, Db1, Dw2, Db2, Dw3, Db3, out, G);
}
